// Model_41188736368876
// MI455X (gfx1250) — matrix operand order repaired and checked by reading, not compiled, not run on hardware
//
#include <hip/hip_runtime.h>
#include <hip/hip_bf16.h>

// ---------------------------------------------------------------------------
// IEGMN forward on gfx1250 (MI455X). All GEMM-shaped work runs through
// v_wmma_f32_16x16x32_f16 (wave32). Per-wave 16x16 output tiles, per-wave LDS
// regions, same-wave LDS ordering via s_wait_dscnt. The equivariant coordinate
// update in the reference is dead code w.r.t. the returned tensor and is
// skipped. f16 data / f32 accumulation everywhere.
// ---------------------------------------------------------------------------

#define HID    64
#define EDIM   16
#define NHEAD  30
#define DHD    8
#define HD     240      // NHEAD*DHD
#define NLIG   512
#define NREC   1024
#define NTOT   1536
#define ELIG   8192
#define EREC   16384
#define NLAYER 8

typedef __attribute__((ext_vector_type(16))) _Float16 v16h;
typedef __attribute__((ext_vector_type(8)))  _Float16 v8h;
typedef __attribute__((ext_vector_type(8)))  float    v8f;

__device__ __forceinline__ v8f wmma_f16(v16h a, v16h b, v8f c) {
  // (neg_a, A, neg_b, B, c_mod, C, reuse_a, reuse_b)
  return __builtin_amdgcn_wmma_f32_16x16x32_f16(false, a, false, b, (short)0, c,
                                                false, false);
}

__device__ __forceinline__ float lrelu(float v, float s) {
  return fmaxf(v, 0.f) + s * fminf(v, 0.f);
}

// A fragment (16x32 f16, row-major source with leading dim lda).
// ISA layout: lanes 0-15 -> row=lane, K {k0+0..7, k0+16..23};
//             lanes 16-31 -> row=lane-16, K {k0+8..15, k0+24..31}.
__device__ __forceinline__ v16h load_a_frag(const _Float16* A, int lda, int k0,
                                            int lane) {
  int row = lane & 15;
  int kb  = k0 + ((lane >> 4) & 1) * 8;
  const _Float16* p = A + row * lda + kb;
  v8h lo = *(const v8h*)(p);
  v8h hi = *(const v8h*)(p + 16);
  v16h a;
#pragma unroll
  for (int i = 0; i < 8; ++i) { a[i] = lo[i]; a[i + 8] = hi[i]; }
  return a;
}

// B fragment (32x16 f16) from a PRE-TRANSPOSED weight [N][Kpad] (ldb = Kpad).
// ISA layout: lanes 0-15 -> col=lane, K {k0+0..7, k0+16..23}; lanes 16-31 -> K {k0+8..15, k0+24..31}.
__device__ __forceinline__ v16h load_b_fragT(const _Float16* Bt, int ldb, int k0,
                                             int lane) {
  int col = lane & 15;
  int kb  = k0 + ((lane & 16) ? 8 : 0);
  const _Float16* p = Bt + col * ldb + kb;
  v8h lo = *(const v8h*)(p);
  v8h hi = *(const v8h*)(p + 16);
  v16h b;
#pragma unroll
  for (int i = 0; i < 8; ++i) { b[i] = lo[i]; b[i + 8] = hi[i]; }
  return b;
}

#define WAVE_LDS_FENCE() asm volatile("s_wait_dscnt 0" ::: "memory")

// ---------------------------------------------------------------------------
// Utility kernels
// ---------------------------------------------------------------------------
__global__ void zero_f32_kernel(float* p, int n) {
  int i = blockIdx.x * blockDim.x + threadIdx.x;
  if (i < n) p[i] = 0.f;
}
__global__ void zero_f16_kernel(_Float16* p, int n) {
  int i = blockIdx.x * blockDim.x + threadIdx.x;
  if (i < n) p[i] = (_Float16)0.f;
}
__global__ void copy_cvt_kernel(const float* __restrict__ src, float* dstf,
                                _Float16* dsth, int n) {
  int i = blockIdx.x * blockDim.x + threadIdx.x;
  if (i >= n) return;
  float v = src[i];
  if (dstf) dstf[i] = v;
  dsth[i] = (_Float16)v;
}
// src f32 [B][K][N] -> dst f16 transposed+padded [B][N][Kp] (rows K..Kp-1 = 0)
__global__ void cvt_wT_kernel(const float* __restrict__ src,
                              _Float16* __restrict__ dst, int B, int K, int N,
                              int Kp) {
  int i = blockIdx.x * blockDim.x + threadIdx.x;
  int total = B * N * Kp;
  if (i >= total) return;
  int b  = i / (N * Kp);
  int r  = i - b * (N * Kp);
  int n  = r / Kp;
  int kp = r - n * Kp;
  float v = (kp < K) ? src[(b * K + kp) * N + n] : 0.f;
  dst[i] = (_Float16)v;
}
__global__ void deg_kernel(const int* __restrict__ ld, const int* __restrict__ rd,
                           float* deg) {
  int i = blockIdx.x * blockDim.x + threadIdx.x;
  if (i < ELIG)            atomicAdd(&deg[ld[i]], 1.f);
  else if (i < ELIG + EREC) atomicAdd(&deg[NLIG + rd[i - ELIG]], 1.f);
}

// ---------------------------------------------------------------------------
// Generic per-wave 16x16-tile GEMM: C = act(A(f16)[M,K] @ Bt(f16)[N,Kp]^T + bias)
// ---------------------------------------------------------------------------
__global__ void gemm16_kernel(const _Float16* __restrict__ A, int lda,
                              const _Float16* __restrict__ Bt, int ldb,
                              const float* __restrict__ bias,
                              _Float16* __restrict__ Ch, float* __restrict__ Cf,
                              int ldc, int mtiles, int ntiles, int K,
                              float slope) {
  int lane = threadIdx.x & 31;
  int wv   = blockIdx.x * (blockDim.x >> 5) + (threadIdx.x >> 5);
  if (wv >= mtiles * ntiles) return;
  int mt = wv % mtiles, nt = wv / mtiles;
  int m0 = mt * 16, n0 = nt * 16;
  v8f acc = {};
  for (int k0 = 0; k0 < K; k0 += 32) {
    v16h a = load_a_frag(A + m0 * lda, lda, k0, lane);
    v16h b = load_b_fragT(Bt + n0 * ldb, ldb, k0, lane);
    acc = wmma_f16(a, b, acc);
  }
  int col = lane & 15;
  int rb  = (lane & 16) ? 8 : 0;
  float bb = bias ? bias[n0 + col] : 0.f;
#pragma unroll
  for (int j = 0; j < 8; ++j) {
    float v = lrelu(acc[j] + bb, slope);
    int idx = (m0 + rb + j) * ldc + n0 + col;
    if (Ch) Ch[idx] = (_Float16)v;
    if (Cf) Cf[idx] = v;
  }
}

// ---------------------------------------------------------------------------
// Edge message MLP + scatter-add aggregation.
// One wave per 16-edge tile; gathers [h_src|h_dst|ef] (144, padded to 160),
// two WMMA GEMM stages, atomic f32 accumulation into agg[dst].
// ---------------------------------------------------------------------------
__global__ void edge_kernel(const _Float16* __restrict__ h16, int hbase,
                            const int* __restrict__ src,
                            const int* __restrict__ dst,
                            const _Float16* __restrict__ ef16,
                            const _Float16* __restrict__ We1t,  // [64][160]
                            const float* __restrict__ be1,
                            const _Float16* __restrict__ We2t,  // [64][64]
                            const float* __restrict__ be2,
                            float* __restrict__ agg) {
  __shared__ _Float16 Ah[4][16][160];
  __shared__ _Float16 Hh[4][16][64];
  __shared__ int sid[4][16], did[4][16];
  int lane = threadIdx.x & 31;
  int wv   = threadIdx.x >> 5;
  int e0   = (blockIdx.x * 4 + wv) * 16;

  if (lane < 16) {
    sid[wv][lane] = src[e0 + lane];
    did[wv][lane] = dst[e0 + lane];
  }
  WAVE_LDS_FENCE();

  for (int i = lane; i < 16 * 160; i += 32) {
    int e = i / 160, f = i - e * 160;
    _Float16 v = (_Float16)0.f;
    if (f < 64)       v = h16[(hbase + sid[wv][e]) * HID + f];
    else if (f < 128) v = h16[(hbase + did[wv][e]) * HID + f - 64];
    else if (f < 144) v = ef16[(e0 + e) * EDIM + f - 128];
    Ah[wv][e][f] = v;
  }
  WAVE_LDS_FENCE();

  int col = lane & 15, rb = (lane & 16) ? 8 : 0;
#pragma unroll
  for (int nt = 0; nt < 4; ++nt) {
    int n0 = nt * 16;
    v8f acc = {};
    for (int k0 = 0; k0 < 160; k0 += 32) {
      v16h a = load_a_frag(&Ah[wv][0][0], 160, k0, lane);
      v16h b = load_b_fragT(We1t + n0 * 160, 160, k0, lane);
      acc = wmma_f16(a, b, acc);
    }
    float bb = be1[n0 + col];
#pragma unroll
    for (int j = 0; j < 8; ++j)
      Hh[wv][rb + j][n0 + col] = (_Float16)lrelu(acc[j] + bb, 0.01f);
  }
  WAVE_LDS_FENCE();

#pragma unroll
  for (int nt = 0; nt < 4; ++nt) {
    int n0 = nt * 16;
    v8f acc = {};
    for (int k0 = 0; k0 < 64; k0 += 32) {
      v16h a = load_a_frag(&Hh[wv][0][0], 64, k0, lane);
      v16h b = load_b_fragT(We2t + n0 * 64, 64, k0, lane);
      acc = wmma_f16(a, b, acc);
    }
    float bb = be2[n0 + col];
#pragma unroll
    for (int j = 0; j < 8; ++j) {
      int d = did[wv][rb + j];
      atomicAdd(&agg[(hbase + d) * HID + n0 + col], acc[j] + bb);
    }
  }
}

// ---------------------------------------------------------------------------
// Flash-style cross attention: one wave = (16 query rows) x (1 head of 30).
// S = QK^T via WMMA (K=8 zero-padded to 32), online softmax with width-16
// shuffle reductions, O += P@V via WMMA (P restaged through LDS into A layout).
// Output written f16 into [rows][256]-padded buffer.
// ---------------------------------------------------------------------------
__global__ void attn_kernel(const _Float16* __restrict__ q,
                            const _Float16* __restrict__ k,
                            const _Float16* __restrict__ v,
                            _Float16* __restrict__ outp, int mtiles, int nkeys) {
  __shared__ _Float16 Ph[4][16][32];
  int lane = threadIdx.x & 31;
  int wv   = threadIdx.x >> 5;
  int tw   = blockIdx.x * 4 + wv;
  int mt   = tw % mtiles;
  int head = tw / mtiles;
  int m0 = mt * 16, h8 = head * DHD;
  int col = lane & 15, rb = (lane & 16) ? 8 : 0;
  const float scale = 0.35355339059f;  // 1/sqrt(8)

  if (lane < 16) {
#pragma unroll
    for (int i = 16; i < 32; ++i) Ph[wv][lane][i] = (_Float16)0.f;  // K pad
  }

  v16h aq = {};
  if (lane < 16) {
    v8h qq = *(const v8h*)(q + (m0 + lane) * HD + h8);
#pragma unroll
    for (int i = 0; i < 8; ++i) aq[i] = qq[i];
  }

  float mrun[8], lrun[8];
  v8f o = {};
#pragma unroll
  for (int j = 0; j < 8; ++j) { mrun[j] = -1e30f; lrun[j] = 0.f; }

  for (int c0 = 0; c0 < nkeys; c0 += 16) {
    v16h bk = {};
    if (lane < 16) {
      v8h kk = *(const v8h*)(k + (c0 + col) * HD + h8);
#pragma unroll
      for (int i = 0; i < 8; ++i) bk[i] = kk[i];
    }
    v8f s0 = {};
    v8f s = wmma_f16(aq, bk, s0);

    float p[8];
#pragma unroll
    for (int j = 0; j < 8; ++j) {
      float sv = s[j] * scale;
      float rm = sv;
      rm = fmaxf(rm, __shfl_xor(rm, 1, 16));
      rm = fmaxf(rm, __shfl_xor(rm, 2, 16));
      rm = fmaxf(rm, __shfl_xor(rm, 4, 16));
      rm = fmaxf(rm, __shfl_xor(rm, 8, 16));
      float mn = fmaxf(mrun[j], rm);
      float pv = __expf(sv - mn);
      float rs = pv;
      rs += __shfl_xor(rs, 1, 16);
      rs += __shfl_xor(rs, 2, 16);
      rs += __shfl_xor(rs, 4, 16);
      rs += __shfl_xor(rs, 8, 16);
      float corr = __expf(mrun[j] - mn);
      lrun[j] = lrun[j] * corr + rs;
      mrun[j] = mn;
      o[j] *= corr;
      p[j] = pv;
    }
#pragma unroll
    for (int j = 0; j < 8; ++j) Ph[wv][rb + j][col] = (_Float16)p[j];
    WAVE_LDS_FENCE();

    v16h ap = load_a_frag(&Ph[wv][0][0], 32, 0, lane);
    v16h bv = {};
    if (col < DHD) {
#pragma unroll
      for (int i = 0; i < 8; ++i) bv[i] = v[(c0 + rb + i) * HD + h8 + col];
    }
    o = wmma_f16(ap, bv, o);
  }

  if (col < DHD) {
#pragma unroll
    for (int j = 0; j < 8; ++j) {
      float ov = o[j] / lrun[j];
      outp[(m0 + rb + j) * 256 + h8 + col] = (_Float16)ov;
    }
  }
}

// ---------------------------------------------------------------------------
// Node update: z = [h | agg/deg | cross]; h += lrelu(z@Wn1+bn1,.01)@Wn2+bn2
// One wave per 16-node tile; updates f32 master + f16 mirror in place.
// ---------------------------------------------------------------------------
__global__ void node_kernel(_Float16* h16, float* h32,
                            const float* __restrict__ agg,
                            const float* __restrict__ deg,
                            const _Float16* __restrict__ cross16,
                            const _Float16* __restrict__ Wn1t,  // [64][192]
                            const float* __restrict__ bn1,
                            const _Float16* __restrict__ Wn2t,  // [64][64]
                            const float* __restrict__ bn2) {
  __shared__ _Float16 Ac[4][16][192];
  __shared__ _Float16 Hh[4][16][64];
  int lane = threadIdx.x & 31;
  int wv   = threadIdx.x >> 5;
  int m0   = (blockIdx.x * 4 + wv) * 16;

  for (int i = lane; i < 16 * 192; i += 32) {
    int r = i / 192, f = i - r * 192;
    int node = m0 + r;
    _Float16 vv;
    if (f < 64) vv = h16[node * HID + f];
    else if (f < 128) {
      float dg = deg[node];
      dg = dg < 1.f ? 1.f : dg;
      vv = (_Float16)(agg[node * HID + f - 64] / dg);
    } else vv = cross16[node * HID + f - 128];
    Ac[wv][r][f] = vv;
  }
  WAVE_LDS_FENCE();

  int col = lane & 15, rb = (lane & 16) ? 8 : 0;
#pragma unroll
  for (int nt = 0; nt < 4; ++nt) {
    int n0 = nt * 16;
    v8f acc = {};
    for (int k0 = 0; k0 < 192; k0 += 32) {
      v16h a = load_a_frag(&Ac[wv][0][0], 192, k0, lane);
      v16h b = load_b_fragT(Wn1t + n0 * 192, 192, k0, lane);
      acc = wmma_f16(a, b, acc);
    }
    float bb = bn1[n0 + col];
#pragma unroll
    for (int j = 0; j < 8; ++j)
      Hh[wv][rb + j][n0 + col] = (_Float16)lrelu(acc[j] + bb, 0.01f);
  }
  WAVE_LDS_FENCE();

#pragma unroll
  for (int nt = 0; nt < 4; ++nt) {
    int n0 = nt * 16;
    v8f acc = {};
    for (int k0 = 0; k0 < 64; k0 += 32) {
      v16h a = load_a_frag(&Hh[wv][0][0], 64, k0, lane);
      v16h b = load_b_fragT(Wn2t + n0 * 64, 64, k0, lane);
      acc = wmma_f16(a, b, acc);
    }
    float bb = bn2[n0 + col];
#pragma unroll
    for (int j = 0; j < 8; ++j) {
      int idx = (m0 + rb + j) * HID + n0 + col;
      float v = acc[j] + bb + h32[idx];
      h32[idx] = v;
      h16[idx] = (_Float16)v;
    }
  }
}

// ---------------------------------------------------------------------------
// Output head: lrelu(h@Wo1+bo1,.2) -> lrelu(@Wo2+bo2,.2) -> @Wo3+bo3
// ---------------------------------------------------------------------------
__global__ void head_kernel(const _Float16* __restrict__ h16,
                            const _Float16* __restrict__ Wo1t,  // [64][64]
                            const float* __restrict__ bo1,
                            const _Float16* __restrict__ Wo2t,  // [16][64]
                            const float* __restrict__ bo2,
                            const float* __restrict__ Wo3,      // [16]
                            const float* __restrict__ bo3,
                            float* __restrict__ out) {
  __shared__ _Float16 H1[4][16][64];
  __shared__ _Float16 H2[4][16][16];
  int lane = threadIdx.x & 31;
  int wv   = threadIdx.x >> 5;
  int m0   = (blockIdx.x * 4 + wv) * 16;
  int col = lane & 15, rb = (lane & 16) ? 8 : 0;

#pragma unroll
  for (int nt = 0; nt < 4; ++nt) {
    int n0 = nt * 16;
    v8f acc = {};
    for (int k0 = 0; k0 < 64; k0 += 32) {
      v16h a = load_a_frag(h16 + m0 * HID, HID, k0, lane);
      v16h b = load_b_fragT(Wo1t + n0 * 64, 64, k0, lane);
      acc = wmma_f16(a, b, acc);
    }
    float bb = bo1[n0 + col];
#pragma unroll
    for (int j = 0; j < 8; ++j)
      H1[wv][rb + j][n0 + col] = (_Float16)lrelu(acc[j] + bb, 0.2f);
  }
  WAVE_LDS_FENCE();

  {
    v8f acc = {};
    for (int k0 = 0; k0 < 64; k0 += 32) {
      v16h a = load_a_frag(&H1[wv][0][0], 64, k0, lane);
      v16h b = load_b_fragT(Wo2t, 64, k0, lane);
      acc = wmma_f16(a, b, acc);
    }
    float bb = bo2[col];
#pragma unroll
    for (int j = 0; j < 8; ++j)
      H2[wv][rb + j][col] = (_Float16)lrelu(acc[j] + bb, 0.2f);
  }
  WAVE_LDS_FENCE();

  if (lane < 16) {
    float s = bo3[0];
#pragma unroll
    for (int c = 0; c < 16; ++c) s += (float)H2[wv][lane][c] * Wo3[c];
    out[m0 + lane] = s;
  }
}

// ---------------------------------------------------------------------------
// Host orchestration
// ---------------------------------------------------------------------------
extern "C" void kernel_launch(void* const* d_in, const int* in_sizes, int n_in,
                              void* d_out, int out_size, void* d_ws,
                              size_t ws_size, hipStream_t stream) {
  (void)in_sizes; (void)n_in; (void)out_size; (void)ws_size;
  const float* lig_h0 = (const float*)d_in[0];
  const float* rec_h0 = (const float*)d_in[1];
  // d_in[2] lig_x: coordinate pathway is dead w.r.t. the output -> skipped
  const float* lig_ef = (const float*)d_in[3];
  const float* rec_ef = (const float*)d_in[4];
  const int* lig_src = (const int*)d_in[5];
  const int* lig_dst = (const int*)d_in[6];
  const int* rec_src = (const int*)d_in[7];
  const int* rec_dst = (const int*)d_in[8];
  const float* We1 = (const float*)d_in[9];
  const float* be1 = (const float*)d_in[10];
  const float* We2 = (const float*)d_in[11];
  const float* be2 = (const float*)d_in[12];
  const float* Wq  = (const float*)d_in[13];
  const float* Wk  = (const float*)d_in[14];
  const float* Wv  = (const float*)d_in[15];
  const float* Wo  = (const float*)d_in[16];
  const float* Wn1 = (const float*)d_in[17];
  const float* bn1 = (const float*)d_in[18];
  const float* Wn2 = (const float*)d_in[19];
  const float* bn2 = (const float*)d_in[20];
  // d_in[21] Wc, d_in[22] bc: dead (coordinate head)
  const float* Wo1 = (const float*)d_in[23];
  const float* bo1 = (const float*)d_in[24];
  const float* Wo2 = (const float*)d_in[25];
  const float* bo2 = (const float*)d_in[26];
  const float* Wo3 = (const float*)d_in[27];
  const float* bo3 = (const float*)d_in[28];

  char* ws = (char*)d_ws;
  size_t off = 0;
  auto carve = [&](size_t bytes) -> void* {
    void* p = ws + off;
    off += (bytes + 255) & ~(size_t)255;
    return p;
  };
  float*    h32     = (float*)carve((size_t)NTOT * HID * 4);
  _Float16* h16     = (_Float16*)carve((size_t)NTOT * HID * 2);
  float*    agg     = (float*)carve((size_t)NTOT * HID * 4);
  float*    deg     = (float*)carve((size_t)NTOT * 4);
  _Float16* cross16 = (_Float16*)carve((size_t)NTOT * HID * 2);
  _Float16* q16     = (_Float16*)carve((size_t)NTOT * HD * 2);
  _Float16* k16     = (_Float16*)carve((size_t)NTOT * HD * 2);
  _Float16* v16p    = (_Float16*)carve((size_t)NTOT * HD * 2);
  _Float16* att16   = (_Float16*)carve((size_t)NTOT * 256 * 2);
  _Float16* efl16   = (_Float16*)carve((size_t)ELIG * EDIM * 2);
  _Float16* efr16   = (_Float16*)carve((size_t)EREC * EDIM * 2);
  _Float16* We1t = (_Float16*)carve((size_t)NLAYER * 64 * 160 * 2);
  _Float16* We2t = (_Float16*)carve((size_t)NLAYER * 64 * 64 * 2);
  _Float16* Wqt  = (_Float16*)carve((size_t)NLAYER * 240 * 64 * 2);
  _Float16* Wkt  = (_Float16*)carve((size_t)NLAYER * 240 * 64 * 2);
  _Float16* Wvt  = (_Float16*)carve((size_t)NLAYER * 240 * 64 * 2);
  _Float16* Wot  = (_Float16*)carve((size_t)NLAYER * 64 * 256 * 2);
  _Float16* Wn1t = (_Float16*)carve((size_t)NLAYER * 64 * 192 * 2);
  _Float16* Wn2t = (_Float16*)carve((size_t)NLAYER * 64 * 64 * 2);
  _Float16* Wo1t = (_Float16*)carve((size_t)64 * 64 * 2);
  _Float16* Wo2t = (_Float16*)carve((size_t)16 * 64 * 2);

  auto g1 = [](int n) { return dim3((unsigned)((n + 255) / 256)); };

  // --- weight conversion (f32 -> f16, transposed [N][Kpad]) ---
  cvt_wT_kernel<<<g1(8 * 64 * 160), 256, 0, stream>>>(We1, We1t, 8, 144, 64, 160);
  cvt_wT_kernel<<<g1(8 * 64 * 64),  256, 0, stream>>>(We2, We2t, 8, 64, 64, 64);
  cvt_wT_kernel<<<g1(8 * 240 * 64), 256, 0, stream>>>(Wq, Wqt, 8, 64, 240, 64);
  cvt_wT_kernel<<<g1(8 * 240 * 64), 256, 0, stream>>>(Wk, Wkt, 8, 64, 240, 64);
  cvt_wT_kernel<<<g1(8 * 240 * 64), 256, 0, stream>>>(Wv, Wvt, 8, 64, 240, 64);
  cvt_wT_kernel<<<g1(8 * 64 * 256), 256, 0, stream>>>(Wo, Wot, 8, 240, 64, 256);
  cvt_wT_kernel<<<g1(8 * 64 * 192), 256, 0, stream>>>(Wn1, Wn1t, 8, 192, 64, 192);
  cvt_wT_kernel<<<g1(8 * 64 * 64),  256, 0, stream>>>(Wn2, Wn2t, 8, 64, 64, 64);
  cvt_wT_kernel<<<g1(64 * 64), 256, 0, stream>>>(Wo1, Wo1t, 1, 64, 64, 64);
  cvt_wT_kernel<<<g1(16 * 64), 256, 0, stream>>>(Wo2, Wo2t, 1, 64, 16, 64);

  // --- state init ---
  copy_cvt_kernel<<<g1(NLIG * HID), 256, 0, stream>>>(lig_h0, h32, h16, NLIG * HID);
  copy_cvt_kernel<<<g1(NREC * HID), 256, 0, stream>>>(
      rec_h0, h32 + NLIG * HID, h16 + NLIG * HID, NREC * HID);
  copy_cvt_kernel<<<g1(ELIG * EDIM), 256, 0, stream>>>(lig_ef, nullptr, efl16, ELIG * EDIM);
  copy_cvt_kernel<<<g1(EREC * EDIM), 256, 0, stream>>>(rec_ef, nullptr, efr16, EREC * EDIM);
  zero_f32_kernel<<<g1(NTOT), 256, 0, stream>>>(deg, NTOT);
  zero_f16_kernel<<<g1(NTOT * 256), 256, 0, stream>>>(att16, NTOT * 256);
  deg_kernel<<<g1(ELIG + EREC), 256, 0, stream>>>(lig_dst, rec_dst, deg);

  for (int l = 0; l < NLAYER; ++l) {
    zero_f32_kernel<<<g1(NTOT * HID), 256, 0, stream>>>(agg, NTOT * HID);
    // edge messages + scatter aggregation
    edge_kernel<<<ELIG / 64, 128, 0, stream>>>(
        h16, 0, lig_src, lig_dst, efl16, We1t + l * 64 * 160, be1 + l * 64,
        We2t + l * 64 * 64, be2 + l * 64, agg);
    edge_kernel<<<EREC / 64, 128, 0, stream>>>(
        h16, NLIG, rec_src, rec_dst, efr16, We1t + l * 64 * 160, be1 + l * 64,
        We2t + l * 64 * 64, be2 + l * 64, agg);
    // Q/K/V projections for both node sets at once (96 m-tiles x 15 n-tiles)
    gemm16_kernel<<<360, 128, 0, stream>>>(h16, HID, Wqt + l * 240 * 64, 64,
                                           nullptr, q16, nullptr, HD, 96, 15,
                                           64, 1.0f);
    gemm16_kernel<<<360, 128, 0, stream>>>(h16, HID, Wkt + l * 240 * 64, 64,
                                           nullptr, k16, nullptr, HD, 96, 15,
                                           64, 1.0f);
    gemm16_kernel<<<360, 128, 0, stream>>>(h16, HID, Wvt + l * 240 * 64, 64,
                                           nullptr, v16p, nullptr, HD, 96, 15,
                                           64, 1.0f);
    // cross attention: lig queries vs rec keys, and vice versa
    attn_kernel<<<240, 128, 0, stream>>>(q16, k16 + NLIG * HD, v16p + NLIG * HD,
                                         att16, NLIG / 16, NREC);
    attn_kernel<<<480, 128, 0, stream>>>(q16 + NLIG * HD, k16, v16p,
                                         att16 + NLIG * 256, NREC / 16, NLIG);
    // cross = attnout @ Wo  (K padded 240->256)
    gemm16_kernel<<<96, 128, 0, stream>>>(att16, 256, Wot + l * 64 * 256, 256,
                                          nullptr, cross16, nullptr, HID, 96, 4,
                                          256, 1.0f);
    // node update (both sets share Wn1/Wn2)
    node_kernel<<<24, 128, 0, stream>>>(h16, h32, agg, deg, cross16,
                                        Wn1t + l * 64 * 192, bn1 + l * 64,
                                        Wn2t + l * 64 * 64, bn2 + l * 64);
  }

  head_kernel<<<8, 128, 0, stream>>>(h16, Wo1t, bo1, Wo2t, bo2, Wo3, bo3,
                                     (float*)d_out);
}
